// MaskedGraphAF_31456340476504
// MI455X (gfx1250) — matrix operand order repaired and checked by reading, not compiled, not run on hardware
//
#include <hip/hip_runtime.h>
#include <math.h>

// ---------------- problem constants ----------------
#define NN    38      // nodes
#define NDIM  4       // node feature dim
#define NEDGE 3       // edge types (adj channels)
#define NHIDC 128
#define BATCH 16
#define EE    703     // N*(N-1)/2
#define RR    741     // N + E
#define GG    (BATCH*RR)   // 11856 masked graphs
#define ROWS_NODE (BATCH*NN)   // 608  (= 38 tiles of 16)
#define ROWS_EDGE (BATCH*EE)   // 11248 (= 703 tiles of 16)
#define BNCNT ((float)(GG*NN)) // rows entering batch-norm stats

typedef __attribute__((ext_vector_type(16))) _Float16 v16h;
typedef __attribute__((ext_vector_type(8)))  _Float16 v8h;
typedef __attribute__((ext_vector_type(8)))  float    v8f;

// d_in flat order (setup_inputs insertion order, params pytree in insertion order)
#define IDX_X      0
#define IDX_ADJ    1
#define IDX_XDEQ   2
#define IDX_ADJDEQ 3
#define IDX_EMBW   4
#define IDX_GCW0   5
#define IDX_GCW1   6
#define IDX_GCW2   7
#define IDX_GCB0   8
#define IDX_GCB1   9
#define IDX_GCB2   10
#define IDX_BNG    11
#define IDX_BNB    12
#define IDX_NODE_ST 13   // 12 x {w1,b1,w2,b2,rescale}
#define IDX_EDGE_ST 73   // 12 x {w1,b1,w2,b2,rescale}

// ---------------- WMMA helpers (CDNA5 layouts, ISA 7.12.2) ----------------
__device__ __forceinline__ v8f wmma16(v16h a, v16h b, v8f c) {
  return __builtin_amdgcn_wmma_f32_16x16x32_f16(false, a, false, b, (short)0, c,
                                                false, false);
}

// A fragment 16x32 f16 from row-major [row][k] (stride in halfs, 16B-aligned rows).
// lane<16: row=lane, K 0..15; lane>=16: row=lane-16, K 16..31
__device__ __forceinline__ v16h load_a(const _Float16* base, int row0, int k0, int stride) {
  const int lane = threadIdx.x & 31;
  const _Float16* p = base + (size_t)(row0 + (lane & 15)) * stride + k0 + ((lane >> 4) << 4);
  v8h lo = *(const v8h*)p;
  v8h hi8 = *(const v8h*)(p + 8);
  v16h a;
#pragma unroll
  for (int t = 0; t < 8; ++t) { a[t] = lo[t]; a[t + 8] = hi8[t]; }
  return a;
}

// B fragment 32x16 f16 from SWIZZLED [col][k] storage (strideK halfs per column,
// 32B-aligned columns). lanes 0-15: col=lane, K=0..15 ; lanes 16-31: K=16..31.
// One aligned 32-byte load per lane.
__device__ __forceinline__ v16h load_b_swz(const _Float16* base, int k0, int col0, int strideK) {
  const int lane = threadIdx.x & 31;
  const _Float16* p = base + (size_t)(col0 + (lane & 15)) * strideK + k0 + ((lane >> 4) << 4);
  return *(const v16h*)p;
}

// ---------------- per-layer ST parameter pointers ----------------
struct STParams {           // index: side*12+l  (side 0=node, 1=edge)
  const float* w1[24];
  const float* b1[24];
  const float* w2[24];
  const float* b2[24];
  const float* rs[24];
};

// ---------------- prep kernels ----------------
// gcw16 swizzled: [layer*2+e][c][k]  (k contiguous)
__global__ void prep_misc_kernel(float* bn_acc, float* M0, _Float16* gcw16,
                                 const float* embw, const float* gcw0,
                                 const float* gcw1, const float* gcw2) {
  int i = blockIdx.x * blockDim.x + threadIdx.x;
  if (i < 256) bn_acc[i] = 0.f;
  if (i < 1024) {  // M0[e][d][c] = sum_t emb_w[d][t] * gc_w0[e][t][c]
    int e = i / 512, d = (i % 512) / 128, c = i % 128;
    float v = 0.f;
#pragma unroll
    for (int t = 0; t < 4; ++t) v += embw[d * 4 + t] * gcw0[(e * 4 + t) * 128 + c];
    M0[(e * 4 + d) * 128 + c] = v;
  }
  if (i < 65536) {
    int le = i / 16384, rem = i % 16384, c = rem / 128, k = rem % 128;
    const float* src = (le < 2) ? gcw1 : gcw2;
    int e = le & 1;
    gcw16[i] = (_Float16)src[(e * 128 + k) * 128 + c];
  }
}

// All flow weights stored swizzled [l][c][k]
__global__ void prep_st_kernel(STParams P, _Float16* nw1, _Float16* ew1,
                               _Float16* nw2p, _Float16* ew2p) {
  int i = blockIdx.x * 256 + threadIdx.x;
  if (i < 12 * 16384) {            // node w1 (128x128) -> f16 [c][k]
    int l = i / 16384, rem = i % 16384, c = rem / 128, k = rem % 128;
    nw1[i] = (_Float16)P.w1[l][k * 128 + c];
  }
  int j = i - 12 * 16384;
  if (j >= 0 && j < 12 * 49152) {  // edge w1 (384x128) -> f16 [c][k], k<384
    int l = j / 49152, rem = j % 49152, c = rem / 384, k = rem % 384;
    ew1[j] = (_Float16)P.w1[12 + l][k * 128 + c];
  }
  int k2 = j - 12 * 49152;
  if (k2 >= 0 && k2 < 12 * 2048) { // node w2 (128x8) -> padded [c<16][k<128]
    int l = k2 / 2048, rem = k2 % 2048, c = rem / 128, kk = rem % 128;
    nw2p[k2] = (_Float16)(c < 8 ? P.w2[l][kk * 8 + c] : 0.f);
  }
  int m = k2 - 12 * 2048;
  if (m >= 0 && m < 12 * 2048) {   // edge w2 (128x6) -> padded [c<16][k<128]
    int l = m / 2048, rem = m % 2048, c = rem / 128, kk = rem % 128;
    ew2p[m] = (_Float16)(c < 6 ? P.w2[12 + l][kk * 6 + c] : 0.f);
  }
}

// ---------------- RGCN kernel: one workgroup per masked graph ----------------
__global__ __launch_bounds__(256)
void rgcn_kernel(const float* __restrict__ x, const float* __restrict__ adj,
                 const float* __restrict__ gcb0, const float* __restrict__ gcb1,
                 const float* __restrict__ gcb2, const float* __restrict__ M0,
                 const _Float16* __restrict__ gcw16,
                 float* __restrict__ graph_sum, float* __restrict__ edge_rows,
                 float* __restrict__ bn_acc) {
  __shared__ __align__(32) _Float16 h16[48 * 128];      // current h, row-major (A side)
  __shared__ __align__(32) _Float16 supT[2][128 * 64];  // support, SWIZZLED [c][k], k<64
  __shared__ __align__(32) _Float16 a16[2][48 * 64];    // masked adjacency, row-major (A side)
  __shared__ float embf[NN * 128];                      // final node_emb (pre-BN)

  const int g = blockIdx.x;
  const int b = g / RR, r = g % RR;
  const int tid = threadIdx.x;
  const int wave = tid >> 5, lane = tid & 31, hi = lane >> 4, c15 = lane & 15;
  const int col = wave * 16 + c15;

  // deterministic mask parameters (recomputed, no mask arrays needed)
  int s = -1, ii = 0, jj = 0, nlim;
  if (r < NN) {
    nlim = r;
  } else {
    s = r - NN;
    int i = 1;
    while ((i * (i + 1)) / 2 <= s) ++i;     // largest i with i(i-1)/2 <= s
    ii = i;
    jj = s - (i * (i - 1)) / 2;
    nlim = i + 1;
  }

  // masked adjacency -> f16 LDS (row-major; used as A operand)
  for (int idx = tid; idx < 2 * 48 * 64; idx += 256) {
    int e = idx / (48 * 64), rem = idx % (48 * 64), n = rem / 64, m = rem % 64;
    float v = 0.f;
    if (n < NN && m < NN && n != m) {
      bool present;
      if (r < NN) present = (n < r) && (m < r);
      else {
        int p = (n > m) ? (n * (n - 1)) / 2 + m : (m * (m - 1)) / 2 + n;
        present = (p < s);
      }
      if (present) v = adj[((size_t)(b * NEDGE + e) * NN + n) * NN + m];
    }
    a16[e][rem] = (_Float16)v;
  }

  // layer-0 support: supT[e][c][k=n] = (xm @ M0_e)[n][c]   (tiny K=4, VALU)
  // k in [nlim,64) explicitly zeroed -> aggregate B reads are always defined.
  for (int idx = tid; idx < 2 * 128 * 64; idx += 256) {
    int e = idx / 8192, rem = idx % 8192, c = rem / 64, n = rem % 64;
    float v = 0.f;
    if (n < nlim) {
      const float* xr = x + (size_t)(b * NN + n) * NDIM;
      const float* m0 = M0 + (size_t)e * 4 * 128 + c;
      v = xr[0] * m0[0] + xr[1] * m0[128] + xr[2] * m0[256] + xr[3] * m0[384];
    }
    supT[e][rem] = (_Float16)v;
  }
  __syncthreads();

  // ---- layer-0 aggregate: h1 = relu(sum_e a_e @ sup_e + b0) ----
  {
    v16h Bagg[2][2];
#pragma unroll
    for (int e = 0; e < 2; ++e)
#pragma unroll
      for (int kt = 0; kt < 2; ++kt)
        Bagg[e][kt] = load_b_swz(&supT[e][0], kt * 32, wave * 16, 64);
    for (int mt = 0; mt < 3; ++mt) {
      float bv = gcb0[col];
      v8f acc;
#pragma unroll
      for (int q = 0; q < 8; ++q) acc[q] = bv;
#pragma unroll
      for (int e = 0; e < 2; ++e)
#pragma unroll
        for (int kt = 0; kt < 2; ++kt) {
          v16h af = load_a(&a16[e][0], mt * 16, kt * 32, 64);
          acc = wmma16(af, Bagg[e][kt], acc);
        }
#pragma unroll
      for (int q = 0; q < 8; ++q) {
        float o = acc[q];
        h16[(size_t)(mt * 16 + hi * 8 + q) * 128 + col] = (_Float16)(o > 0.f ? o : 0.f);
      }
    }
  }

  // ---- layers 1 and 2 ----
  for (int layer = 0; layer < 2; ++layer) {
    __syncthreads();
    // support: sup_e = h @ W_e ; C written transposed (contiguous v8h stores)
    {
      const _Float16* Wl = gcw16 + (size_t)layer * 2 * 16384;
      v16h Bsup[2][4];
#pragma unroll
      for (int e = 0; e < 2; ++e)
#pragma unroll
        for (int kt = 0; kt < 4; ++kt)
          Bsup[e][kt] = load_b_swz(Wl + (size_t)e * 16384, kt * 32, wave * 16, 128);
      for (int mt = 0; mt < 3; ++mt) {
        v16h A[4];
#pragma unroll
        for (int kt = 0; kt < 4; ++kt) A[kt] = load_a(h16, mt * 16, kt * 32, 128);
#pragma unroll
        for (int e = 0; e < 2; ++e) {
          v8f acc;
#pragma unroll
          for (int q = 0; q < 8; ++q) acc[q] = 0.f;
#pragma unroll
          for (int kt = 0; kt < 4; ++kt) acc = wmma16(A[kt], Bsup[e][kt], acc);
          v8h st;
#pragma unroll
          for (int q = 0; q < 8; ++q) st[q] = (_Float16)acc[q];
          *(v8h*)&supT[e][(size_t)col * 64 + mt * 16 + hi * 8] = st;
        }
      }
    }
    __syncthreads();
    // aggregate: out = sum_e a_e @ sup_e + b ; relu except last layer
    const float* gb = (layer == 0) ? gcb1 : gcb2;
    const bool last = (layer == 1);
    {
      v16h Bagg[2][2];
#pragma unroll
      for (int e = 0; e < 2; ++e)
#pragma unroll
        for (int kt = 0; kt < 2; ++kt)
          Bagg[e][kt] = load_b_swz(&supT[e][0], kt * 32, wave * 16, 64);
      for (int mt = 0; mt < 3; ++mt) {
        float bv = gb[col];
        v8f acc;
#pragma unroll
        for (int q = 0; q < 8; ++q) acc[q] = bv;
#pragma unroll
        for (int e = 0; e < 2; ++e)
#pragma unroll
          for (int kt = 0; kt < 2; ++kt) {
            v16h af = load_a(&a16[e][0], mt * 16, kt * 32, 64);
            acc = wmma16(af, Bagg[e][kt], acc);
          }
#pragma unroll
        for (int q = 0; q < 8; ++q) {
          int row = mt * 16 + hi * 8 + q;
          float o = acc[q];
          if (!last) {
            h16[(size_t)row * 128 + col] = (_Float16)(o > 0.f ? o : 0.f);
          } else if (row < NN) {
            embf[row * 128 + col] = o;
          }
        }
      }
    }
  }
  __syncthreads();

  // per-graph sums, BN stat atomics, and the 2 gathered rows for edge graphs
  if (tid < 128) {
    int c = tid;
    float sum = 0.f, sq = 0.f;
    for (int n = 0; n < NN; ++n) {
      float v = embf[n * 128 + c];
      sum += v; sq += v * v;
    }
    graph_sum[(size_t)g * 128 + c] = sum;
    atomicAdd(&bn_acc[c], sum);
    atomicAdd(&bn_acc[128 + c], sq);
    if (r >= NN) edge_rows[(size_t)(b * EE + s) * 256 + c] = embf[ii * 128 + c];
  } else {
    int c = tid - 128;
    if (r >= NN) edge_rows[(size_t)(b * EE + s) * 256 + 128 + c] = embf[jj * 128 + c];
  }
}

// ---------------- finalize: BN affine + build emb_node / emb_edge (f16) ------
__global__ __launch_bounds__(128)
void finalize_kernel(const float* __restrict__ bn_acc, const float* __restrict__ gamma,
                     const float* __restrict__ beta, const float* __restrict__ graph_sum,
                     const float* __restrict__ edge_rows,
                     _Float16* __restrict__ emb_node, _Float16* __restrict__ emb_edge) {
  const int g = blockIdx.x, c = threadIdx.x;
  const int b = g / RR, r = g % RR;
  float mean = bn_acc[c] / BNCNT;
  float var  = bn_acc[128 + c] / BNCNT - mean * mean;
  float sc   = rsqrtf(var + 1e-5f) * gamma[c];
  float be   = beta[c];
  float S = graph_sum[(size_t)g * 128 + c];
  float gaff = (S - (float)NN * mean) * sc + (float)NN * be;  // sum of normalized rows
  if (r < NN) {
    emb_node[(size_t)(b * NN + r) * 128 + c] = (_Float16)gaff;
  } else {
    int s = r - NN;
    size_t eb = (size_t)(b * EE + s);
    float ri = edge_rows[eb * 256 + c];
    float rj = edge_rows[eb * 256 + 128 + c];
    _Float16* dst = emb_edge + eb * 384;
    dst[c]       = (_Float16)((ri - mean) * sc + be);
    dst[128 + c] = (_Float16)((rj - mean) * sc + be);
    dst[256 + c] = (_Float16)gaff;
  }
}

// ---------------- flow kernel: 16 rows per block, 12 coupling layers ---------
__global__ __launch_bounds__(256)
void flow_kernel(const _Float16* __restrict__ emb, int K,
                 const _Float16* __restrict__ w1s, const _Float16* __restrict__ w2s,
                 STParams P, int side, int out_dim,
                 const float* __restrict__ deq, float* __restrict__ zout,
                 float* __restrict__ lj) {
  __shared__ __align__(32) _Float16 hid16[16 * 128];
  __shared__ float ztile[16 * 16];
  const int tid = threadIdx.x, wave = tid >> 5, lane = tid & 31;
  const int hi = lane >> 4, c15 = lane & 15;
  const int col = wave * 16 + c15;
  const int row0 = blockIdx.x * 16;
  const int nk = K / 32;

  // embedding A-fragments are constant over all 12 layers -> keep in VGPRs
  v16h afr[12];
  for (int kt = 0; kt < nk; ++kt) afr[kt] = load_a(emb, row0, kt * 32, K);

  float xr[4] = {0.f, 0.f, 0.f, 0.f};
  float ljacc = 0.f;
  if (tid < 16) {
    int row = row0 + tid;
    for (int d = 0; d < out_dim; ++d) xr[d] = deq[(size_t)row * out_dim + d];
  }

  for (int l = 0; l < 12; ++l) {
    const int pi = side * 12 + l;
    // hid = tanh(emb @ w1 + b1)   (per wave: one 16x16 tile of the 16x128 hidden)
    float bv = P.b1[pi][col];
    v8f acc;
#pragma unroll
    for (int q = 0; q < 8; ++q) acc[q] = bv;
    const _Float16* w1 = w1s + (size_t)l * K * 128;   // swizzled [c][k]
    for (int kt = 0; kt < nk; ++kt) {
      v16h bf = load_b_swz(w1, kt * 32, wave * 16, K);
      acc = wmma16(afr[kt], bf, acc);
    }
#pragma unroll
    for (int q = 0; q < 8; ++q)
      hid16[(size_t)(hi * 8 + q) * 128 + col] = (_Float16)tanhf(acc[q]);
    __syncthreads();

    // z = hid @ w2p + b2  (one 16x16 tile; wave 0 only, full EXEC within wave)
    if (wave == 0) {
      float b2v = (c15 < 2 * out_dim) ? P.b2[pi][c15] : 0.f;
      v8f zacc;
#pragma unroll
      for (int q = 0; q < 8; ++q) zacc[q] = b2v;
      const _Float16* w2 = w2s + (size_t)l * 2048;    // swizzled [c<16][k<128]
#pragma unroll
      for (int kt = 0; kt < 4; ++kt) {
        v16h af2 = load_a(hid16, 0, kt * 32, 128);
        v16h bf2 = load_b_swz(w2, kt * 32, 0, 128);
        zacc = wmma16(af2, bf2, zacc);
      }
#pragma unroll
      for (int q = 0; q < 8; ++q) ztile[(hi * 8 + q) * 16 + c15] = zacc[q];
    }
    __syncthreads();

    // coupling update + log-Jacobian (state lives in registers of threads 0..15)
    if (tid < 16) {
      float rs = expf(*P.rs[pi]);
      for (int d = 0; d < out_dim; ++d) {
        float z0 = ztile[tid * 16 + 2 * d];
        float z1 = ztile[tid * 16 + 2 * d + 1];
        float sg = rs / (1.f + expf(-(z0 + 2.f)));   // sigmoid(z0+SHIFT)*exp(rescale)
        xr[d] = xr[d] * sg + z1;
        ljacc += logf(fabsf(sg) + 1e-20f);
      }
    }
    __syncthreads();
  }

  if (tid < 16) {
    int row = row0 + tid;
    for (int d = 0; d < out_dim; ++d) zout[(size_t)row * out_dim + d] = xr[d];
    lj[row] = ljacc;
  }
}

// ---------------- per-batch logdet reduction ----------------
__global__ void logdet_kernel(const float* __restrict__ ljn, const float* __restrict__ lje,
                              float* __restrict__ out_ldn, float* __restrict__ out_lde) {
  int t = threadIdx.x;
  if (t < 16) {
    float sum = 0.f;
    for (int n = 0; n < NN; ++n) sum += ljn[t * NN + n];
    out_ldn[t] = sum;
  } else if (t < 32) {
    int b = t - 16;
    float sum = 0.f;
    for (int e = 0; e < EE; ++e) sum += lje[(size_t)b * EE + e];
    out_lde[b] = sum;
  }
}

// ---------------- host launcher ----------------
extern "C" void kernel_launch(void* const* d_in, const int* in_sizes, int n_in,
                              void* d_out, int out_size, void* d_ws, size_t ws_size,
                              hipStream_t stream) {
  (void)in_sizes; (void)n_in; (void)out_size; (void)ws_size;

  // ---- workspace carve (256B aligned) ----
  char* w = (char*)d_ws;
  auto carve = [&](size_t bytes) -> void* {
    void* p = (void*)w;
    w += (bytes + 255) & ~(size_t)255;
    return p;
  };
  float*    bn_acc    = (float*)carve(256 * 4);
  float*    M0        = (float*)carve(1024 * 4);
  _Float16* gcw16     = (_Float16*)carve(65536 * 2);
  float*    graph_sum = (float*)carve((size_t)GG * 128 * 4);
  float*    edge_rows = (float*)carve((size_t)BATCH * EE * 256 * 4);
  _Float16* emb_node  = (_Float16*)carve((size_t)ROWS_NODE * 128 * 2);
  _Float16* emb_edge  = (_Float16*)carve((size_t)ROWS_EDGE * 384 * 2);
  _Float16* nw1       = (_Float16*)carve((size_t)12 * 128 * 128 * 2);
  _Float16* ew1       = (_Float16*)carve((size_t)12 * 384 * 128 * 2);
  _Float16* nw2p      = (_Float16*)carve((size_t)12 * 128 * 16 * 2);
  _Float16* ew2p      = (_Float16*)carve((size_t)12 * 128 * 16 * 2);
  float*    lj_node   = (float*)carve((size_t)ROWS_NODE * 4);
  float*    lj_edge   = (float*)carve((size_t)ROWS_EDGE * 4);

  // ---- per-layer ST parameter pointers ----
  STParams P;
  for (int l = 0; l < 12; ++l) {
    P.w1[l]      = (const float*)d_in[IDX_NODE_ST + 5 * l + 0];
    P.b1[l]      = (const float*)d_in[IDX_NODE_ST + 5 * l + 1];
    P.w2[l]      = (const float*)d_in[IDX_NODE_ST + 5 * l + 2];
    P.b2[l]      = (const float*)d_in[IDX_NODE_ST + 5 * l + 3];
    P.rs[l]      = (const float*)d_in[IDX_NODE_ST + 5 * l + 4];
    P.w1[12 + l] = (const float*)d_in[IDX_EDGE_ST + 5 * l + 0];
    P.b1[12 + l] = (const float*)d_in[IDX_EDGE_ST + 5 * l + 1];
    P.w2[12 + l] = (const float*)d_in[IDX_EDGE_ST + 5 * l + 2];
    P.b2[12 + l] = (const float*)d_in[IDX_EDGE_ST + 5 * l + 3];
    P.rs[12 + l] = (const float*)d_in[IDX_EDGE_ST + 5 * l + 4];
  }

  float* out = (float*)d_out;
  float* z_node = out;                       // B*N*ND = 2432
  float* z_edge = out + BATCH * NN * NDIM;   // B*E*NE = 33744
  float* ld_n   = out + BATCH * NN * NDIM + BATCH * EE * NEDGE;
  float* ld_e   = ld_n + BATCH;

  // 1) prep: zero BN atomics, fold emb_w into layer-0 weights, swizzled f16 weights
  prep_misc_kernel<<<256, 256, 0, stream>>>(
      bn_acc, M0, gcw16, (const float*)d_in[IDX_EMBW], (const float*)d_in[IDX_GCW0],
      (const float*)d_in[IDX_GCW1], (const float*)d_in[IDX_GCW2]);
  prep_st_kernel<<<3264, 256, 0, stream>>>(P, nw1, ew1, nw2p, ew2p);

  // 2) RGCN over all masked graphs (the WMMA-heavy stage)
  rgcn_kernel<<<GG, 256, 0, stream>>>(
      (const float*)d_in[IDX_X], (const float*)d_in[IDX_ADJ],
      (const float*)d_in[IDX_GCB0], (const float*)d_in[IDX_GCB1],
      (const float*)d_in[IDX_GCB2], M0, gcw16, graph_sum, edge_rows, bn_acc);

  // 3) batch-norm finalize + embedding assembly
  finalize_kernel<<<GG, 128, 0, stream>>>(
      bn_acc, (const float*)d_in[IDX_BNG], (const float*)d_in[IDX_BNB],
      graph_sum, edge_rows, emb_node, emb_edge);

  // 4) 12-layer affine coupling flows (node and edge branches)
  flow_kernel<<<ROWS_NODE / 16, 256, 0, stream>>>(
      emb_node, 128, nw1, nw2p, P, /*side=*/0, /*out_dim=*/NDIM,
      (const float*)d_in[IDX_XDEQ], z_node, lj_node);
  flow_kernel<<<ROWS_EDGE / 16, 256, 0, stream>>>(
      emb_edge, 384, ew1, ew2p, P, /*side=*/1, /*out_dim=*/NEDGE,
      (const float*)d_in[IDX_ADJDEQ], z_edge, lj_edge);

  // 5) per-batch log-det reductions
  logdet_kernel<<<1, 64, 0, stream>>>(lj_node, lj_edge, ld_n, ld_e);
}
